// FlashAttention_3624952398462
// MI455X (gfx1250) — matrix operand order repaired and checked by reading, not compiled, not run on hardware
//
#include <hip/hip_runtime.h>

// MI455X / gfx1250, wave32. Compute-bound attention block: all GEMM-shaped work
// goes through v_wmma_f32_16x16x32_bf16 (f32 accumulate); flash-attention K/V
// tiles are double-buffered in LDS via GLOBAL_LOAD_ASYNC_TO_LDS_B128
// (ASYNCcnt-tracked async copies, hidden behind the WMMA stream).
//
// Workspace layout (needs ~80 MB):
//   [0 .. 16MB)   Q  bf16 [B,H,T,D]
//   [16 .. 32MB)  K  bf16 [B,H,T,D]
//   [32 .. 48MB)  V  bf16 [B,H,T,D]
//   [48 .. 80MB)  attn-out f32 [B,T,C]

typedef __bf16 bf16;
typedef __attribute__((ext_vector_type(16))) __bf16 v16bf;
typedef __attribute__((ext_vector_type(8)))  float  v8f;

#define B_DIM 2
#define T_DIM 2048
#define C_DIM 2048
#define H_DIM 16
#define D_DIM 128

__device__ __forceinline__ v8f wmma_bf16(v16bf a, v16bf b, v8f c) {
  // (neg_a, A, neg_b, B, c_mod, C, reuse_a, reuse_b)
  return __builtin_amdgcn_wmma_f32_16x16x32_bf16(false, a, false, b, (short)0, c,
                                                 false, false);
}

__device__ __forceinline__ v8f vzero8() {
  v8f z;
#pragma unroll
  for (int i = 0; i < 8; ++i) z[i] = 0.0f;
  return z;
}

// 16-bit A-matrix 16x32 layout: element e of the per-lane v16bf -> K offset.
// lanes 0-15: VGPR0-3 K=0..7, VGPR4-7 K=16..23; lanes 16-31: +8.
__device__ __forceinline__ int kkA(int e, int halfsel) {
  return ((e & 8) << 1) + halfsel * 8 + (e & 7);
}

// Async global->LDS copy of one 32x128 bf16 tile (8 KB = 512 x 16B chunks),
// 4 chunks per thread of a 128-thread block. Tracked by ASYNCcnt.
__device__ __forceinline__ void async_copy_tile_8k(const bf16* gsrc,
                                                   unsigned lds_base, int tid) {
  unsigned long long g = (unsigned long long)(const void*)gsrc;
#pragma unroll
  for (int j = 0; j < 4; ++j) {
    int chunk = tid + j * 128;
    unsigned lds_addr = lds_base + (unsigned)chunk * 16u;
    unsigned long long ga = g + (unsigned long long)chunk * 16ull;
    asm volatile("global_load_async_to_lds_b128 %0, %1, off"
                 :: "v"(lds_addr), "v"(ga)
                 : "memory");
  }
}

// ---------------------------------------------------------------------------
// Tiled GEMM: C[M x N] = A[M x K] * B[K x N], A/B f32 in memory, bf16 WMMA.
// Block = 128 threads (4 waves), 64x64 tile, each wave does 2x2 16x16 tiles.
// MODE 0: plain f32 output (row-major M x N).
// MODE 1: QKV epilogue -> scatter bf16 into head-major Q/K/V [B,H,T,D].
// (Synchronous staging: the f32->bf16 conversion must pass through VGPRs.)
// ---------------------------------------------------------------------------
template <int MODE>
__global__ __launch_bounds__(128) void gemm_bf16_kernel(
    const float* __restrict__ A, const float* __restrict__ B,
    float* __restrict__ Cout, int N, int K,
    bf16* __restrict__ qout, bf16* __restrict__ kout, bf16* __restrict__ vout)
{
  __shared__ alignas(16) bf16 ldsA[64][32];
  __shared__ alignas(16) bf16 ldsB[32][64];

  const int tid = threadIdx.x;
  const int lane = tid & 31, wave = tid >> 5;
  const int halfsel = lane >> 4, lm = lane & 15;
  const int m0w = (wave >> 1) * 32, n0w = (wave & 1) * 32;
  const int gm0 = blockIdx.y * 64, gn0 = blockIdx.x * 64;

  v8f acc[2][2];
#pragma unroll
  for (int i = 0; i < 2; ++i)
#pragma unroll
    for (int j = 0; j < 2; ++j) acc[i][j] = vzero8();

  for (int k0 = 0; k0 < K; k0 += 32) {
    for (int idx = tid; idx < 64 * 32; idx += 128) {
      int r = idx >> 5, c = idx & 31;
      ldsA[r][c] = (bf16)A[(size_t)(gm0 + r) * K + (k0 + c)];
    }
    for (int idx = tid; idx < 32 * 64; idx += 128) {
      int r = idx >> 6, c = idx & 63;
      ldsB[r][c] = (bf16)B[(size_t)(k0 + r) * N + (gn0 + c)];
    }
    __syncthreads();

    v16bf af[2], bfr[2];
#pragma unroll
    for (int mt = 0; mt < 2; ++mt) {
      const bf16* row = &ldsA[m0w + mt * 16 + lm][0];
#pragma unroll
      for (int e = 0; e < 16; ++e) af[mt][e] = row[kkA(e, halfsel)];
    }
#pragma unroll
    for (int nt = 0; nt < 2; ++nt) {
#pragma unroll
      for (int e = 0; e < 16; ++e)
        bfr[nt][e] = ldsB[kkA(e, halfsel)][n0w + nt * 16 + lm];
    }
#pragma unroll
    for (int mt = 0; mt < 2; ++mt)
#pragma unroll
      for (int nt = 0; nt < 2; ++nt)
        acc[mt][nt] = wmma_bf16(af[mt], bfr[nt], acc[mt][nt]);
    __syncthreads();
  }

#pragma unroll
  for (int mt = 0; mt < 2; ++mt)
#pragma unroll
    for (int nt = 0; nt < 2; ++nt)
#pragma unroll
      for (int r = 0; r < 8; ++r) {
        int grow = gm0 + m0w + mt * 16 + halfsel * 8 + r;
        int gcol = gn0 + n0w + nt * 16 + lm;
        float v = acc[mt][nt][r];
        if (MODE == 0) {
          Cout[(size_t)grow * N + gcol] = v;
        } else {
          int b = grow >> 11, t = grow & (T_DIM - 1);
          int sec = gcol >> 11, c = gcol & (C_DIM - 1);
          int h = c >> 7, d = c & (D_DIM - 1);
          bf16* dst = (sec == 0) ? qout : ((sec == 1) ? kout : vout);
          dst[(((size_t)b * H_DIM + h) * T_DIM + t) * D_DIM + d] = (bf16)v;
        }
      }
}

// ---------------------------------------------------------------------------
// RoPE in place on Q and K (bf16, [B,H,T,D]); one thread per rotation pair.
// ---------------------------------------------------------------------------
__global__ __launch_bounds__(256) void rope_kernel(bf16* __restrict__ Q,
                                                   bf16* __restrict__ K)
{
  int tid = blockIdx.x * 256 + threadIdx.x;   // B*H*T*(D/2) threads
  int i  = tid & 63;
  int t  = (tid >> 6) & (T_DIM - 1);
  int bh = tid >> 17;                          // 0 .. B*H-1
  size_t base = ((size_t)bh * T_DIM + t) * D_DIM;

  float inv = expf(-(float)(2 * i) * (1.0f / (float)D_DIM) * logf(10000.0f));
  float ang = (float)t * inv;
  float c = cosf(ang), s = sinf(ang);

  float q1 = (float)Q[base + i], q2 = (float)Q[base + i + 64];
  Q[base + i]      = (bf16)(q1 * c - q2 * s);
  Q[base + i + 64] = (bf16)(q1 * s + q2 * c);
  float k1 = (float)K[base + i], k2 = (float)K[base + i + 64];
  K[base + i]      = (bf16)(k1 * c - k2 * s);
  K[base + i + 64] = (bf16)(k1 * s + k2 * c);
}

// ---------------------------------------------------------------------------
// Flash attention (causal, online softmax). Block = 128 threads (4 waves),
// 64 q-rows per block (16 per wave), kv tiles of 32, D=128.
// K/V tiles double-buffered in LDS via async global->LDS copies (ASYNCcnt):
// prefetch tile i+1 while 16 WMMAs consume tile i.
// Output written f32 in [B, T, C] (heads re-interleaved) for the out-proj.
// ---------------------------------------------------------------------------
__global__ __launch_bounds__(128) void fattn_kernel(
    const bf16* __restrict__ Q, const bf16* __restrict__ K,
    const bf16* __restrict__ V, float* __restrict__ O)
{
  __shared__ alignas(16) bf16 ldsK[2][32][128];   // 2 x 8 KB
  __shared__ alignas(16) bf16 ldsV[2][32][128];   // 2 x 8 KB
  __shared__ alignas(16) bf16 ldsP[4][16][32];    // 4 KB

  const int bh = blockIdx.x;            // b*H + h
  const int q0 = blockIdx.y * 64;
  const int tid = threadIdx.x, lane = tid & 31, wave = tid >> 5;
  const int halfsel = lane >> 4, lm = lane & 15;
  const int qw = q0 + wave * 16;

  const bf16* Qb = Q + (size_t)bh * T_DIM * D_DIM;
  const bf16* Kb = K + (size_t)bh * T_DIM * D_DIM;
  const bf16* Vb = V + (size_t)bh * T_DIM * D_DIM;

  const unsigned ldsK_base[2] = {
      (unsigned)(unsigned long long)(const void*)&ldsK[0][0][0],
      (unsigned)(unsigned long long)(const void*)&ldsK[1][0][0]};
  const unsigned ldsV_base[2] = {
      (unsigned)(unsigned long long)(const void*)&ldsV[0][0][0],
      (unsigned)(unsigned long long)(const void*)&ldsV[1][0][0]};

  // Resident Q fragments: 4 x (16x32) A-fragments over D=128.
  v16bf qf[4];
#pragma unroll
  for (int d0 = 0; d0 < 4; ++d0)
#pragma unroll
    for (int e = 0; e < 16; ++e)
      qf[d0][e] = Qb[(size_t)(qw + lm) * D_DIM + d0 * 32 + kkA(e, halfsel)];

  v8f oacc[8];
#pragma unroll
  for (int dt = 0; dt < 8; ++dt) oacc[dt] = vzero8();
  float mrow[8], lrow[8];
#pragma unroll
  for (int r = 0; r < 8; ++r) { mrow[r] = -1e30f; lrow[r] = 0.0f; }

  const float scale = 0.088388347648318447f;   // 1/sqrt(128)
  const int ntiles = (q0 + 64) / 32;           // causal bound, uniform per block

  // Prefetch tile 0 into buffer 0 (8 async b128 per thread: 4 K + 4 V).
  async_copy_tile_8k(Kb, ldsK_base[0], tid);
  async_copy_tile_8k(Vb, ldsV_base[0], tid);

  for (int it = 0; it < ntiles; ++it) {
    const int buf = it & 1;
    const int kv0 = it * 32;

    if (it + 1 < ntiles) {
      // Prefetch next tile into the other buffer, then wait until only the
      // just-issued 8 copies remain in flight: in-order completion means the
      // current tile's batch has fully landed in LDS.
      async_copy_tile_8k(Kb + (size_t)(kv0 + 32) * D_DIM, ldsK_base[buf ^ 1], tid);
      async_copy_tile_8k(Vb + (size_t)(kv0 + 32) * D_DIM, ldsV_base[buf ^ 1], tid);
      asm volatile("s_wait_asynccnt 0x8" ::: "memory");
    } else {
      asm volatile("s_wait_asynccnt 0x0" ::: "memory");
    }
    __syncthreads();   // current tile visible to all waves

    // S = Q * K^T for the wave's 16x32 block of scores
    v8f sacc[2];
    sacc[0] = vzero8(); sacc[1] = vzero8();
#pragma unroll
    for (int d0 = 0; d0 < 4; ++d0) {
#pragma unroll
      for (int nt = 0; nt < 2; ++nt) {
        v16bf bfr;  // B[d][j] = K[kv0+j][d]; lane -> j, element -> d
#pragma unroll
        for (int e = 0; e < 16; ++e)
          bfr[e] = ldsK[buf][nt * 16 + lm][d0 * 32 + kkA(e, halfsel)];
        sacc[nt] = wmma_bf16(qf[d0], bfr, sacc[nt]);
      }
    }

    // online softmax (C/D layout: row = 8*halfsel + r, col = lm / lm+16)
#pragma unroll
    for (int r = 0; r < 8; ++r) {
      int qrow = qw + halfsel * 8 + r;
      float v0 = sacc[0][r] * scale;
      float v1 = sacc[1][r] * scale;
      if (kv0 + lm      > qrow) v0 = -1e30f;
      if (kv0 + 16 + lm > qrow) v1 = -1e30f;
      float mx = fmaxf(v0, v1);
#pragma unroll
      for (int off = 1; off < 16; off <<= 1)
        mx = fmaxf(mx, __shfl_xor(mx, off));    // stays within 16-lane half
      float mnew = fmaxf(mrow[r], mx);
      float p0 = __expf(v0 - mnew);
      float p1 = __expf(v1 - mnew);
      float rs = p0 + p1;
#pragma unroll
      for (int off = 1; off < 16; off <<= 1)
        rs += __shfl_xor(rs, off);
      float alpha = __expf(mrow[r] - mnew);
      lrow[r] = lrow[r] * alpha + rs;
      mrow[r] = mnew;
#pragma unroll
      for (int dt = 0; dt < 8; ++dt) oacc[dt][r] *= alpha;
      int m = halfsel * 8 + r;
      ldsP[wave][m][lm]      = (bf16)p0;        // re-layout P via LDS
      ldsP[wave][m][lm + 16] = (bf16)p1;
    }
    __syncthreads();

    // O += P(16x32) * V(32x128)
    v16bf pf;
#pragma unroll
    for (int e = 0; e < 16; ++e) pf[e] = ldsP[wave][lm][kkA(e, halfsel)];
#pragma unroll
    for (int dt = 0; dt < 8; ++dt) {
      v16bf vf;
#pragma unroll
      for (int e = 0; e < 16; ++e)
        vf[e] = ldsV[buf][kkA(e, halfsel)][dt * 16 + lm];
      oacc[dt] = wmma_bf16(pf, vf, oacc[dt]);
    }
    __syncthreads();   // all reads of `buf` done before it is refilled
  }

  // normalize + write [B, T, C] with heads interleaved
  const int b = bh >> 4, h = bh & 15;
#pragma unroll
  for (int r = 0; r < 8; ++r) {
    float invl = 1.0f / lrow[r];
    int t = qw + halfsel * 8 + r;
#pragma unroll
    for (int dt = 0; dt < 8; ++dt) {
      int cc = h * D_DIM + dt * 16 + lm;
      O[((size_t)b * T_DIM + t) * C_DIM + cc] = oacc[dt][r] * invl;
    }
  }
}

// ---------------------------------------------------------------------------
extern "C" void kernel_launch(void* const* d_in, const int* in_sizes, int n_in,
                              void* d_out, int out_size, void* d_ws, size_t ws_size,
                              hipStream_t stream) {
  (void)in_sizes; (void)n_in; (void)out_size; (void)ws_size;
  const float* x     = (const float*)d_in[0];   // [B,T,C]
  const float* w_qkv = (const float*)d_in[1];   // [C,3C]
  const float* w_out = (const float*)d_in[2];   // [C,C]
  float* out = (float*)d_out;                   // [B,T,C]

  char* ws = (char*)d_ws;
  const size_t HEADBUF = (size_t)B_DIM * H_DIM * T_DIM * D_DIM * sizeof(bf16); // 16 MB
  bf16*  Qb   = (bf16*)(ws);
  bf16*  Kb   = (bf16*)(ws + HEADBUF);
  bf16*  Vb   = (bf16*)(ws + 2 * HEADBUF);
  float* attn = (float*)(ws + 3 * HEADBUF);     // 32 MB, [B,T,C]

  // 1) QKV projection: (B*T x C) * (C x 3C), scatter bf16 into Q/K/V [B,H,T,D]
  gemm_bf16_kernel<1><<<dim3(3 * C_DIM / 64, B_DIM * T_DIM / 64), 128, 0, stream>>>(
      x, w_qkv, nullptr, 3 * C_DIM, C_DIM, Qb, Kb, Vb);

  // 2) RoPE in place on Q, K
  rope_kernel<<<(B_DIM * H_DIM * T_DIM * (D_DIM / 2)) / 256, 256, 0, stream>>>(Qb, Kb);

  // 3) Causal flash attention -> attn [B,T,C] f32
  fattn_kernel<<<dim3(B_DIM * H_DIM, T_DIM / 64), 128, 0, stream>>>(Qb, Kb, Vb, attn);

  // 4) Output projection: (B*T x C) * (C x C)
  gemm_bf16_kernel<0><<<dim3(C_DIM / 64, B_DIM * T_DIM / 64), 128, 0, stream>>>(
      attn, w_out, out, C_DIM, C_DIM, nullptr, nullptr, nullptr);
}
